// GCN4Rec_7705171329593
// MI455X (gfx1250) — matrix operand order repaired and checked by reading, not compiled, not run on hardware
//
#include <hip/hip_runtime.h>
#include <hip/hip_bf16.h>
#include <math.h>

typedef __attribute__((ext_vector_type(16))) _Float16 v16h;
typedef __attribute__((ext_vector_type(8)))  _Float16 v8h;
typedef __attribute__((ext_vector_type(8)))  float    v8f;

#define DIM 64
#define TPB 256

// ---------------- utility kernels ----------------

__global__ void fill_zero(float* __restrict__ p, long n) {
  long t = (long)blockIdx.x * blockDim.x + threadIdx.x;
  if (t < n) p[t] = 0.0f;
}

__global__ void deg_count(const int* __restrict__ dst, float* __restrict__ deg, int E) {
  int t = blockIdx.x * blockDim.x + threadIdx.x;
  if (t < E) atomicAdd(&deg[dst[t]], 1.0f);
}

__global__ void deg_to_dinv(float* __restrict__ deg, int n) {
  int t = blockIdx.x * blockDim.x + threadIdx.x;
  if (t < n) deg[t] = rsqrtf(deg[t] + 1.0f);   // +1 self loop, deg>=1
}

// W [K,N] f32 row-major  ->  Wt [N,K] f16 row-major (B-fragment friendly)
__global__ void transpose_to_f16(const float* __restrict__ W, _Float16* __restrict__ Wt,
                                 int K, int N) {
  int t = blockIdx.x * blockDim.x + threadIdx.x;
  if (t >= K * N) return;
  int k = t / N, n = t - k * N;
  Wt[(long)n * K + k] = (_Float16)W[t];
}

// max_norm1 over rows of dim 64: one wave per row, float2 per lane; f16 output.
__global__ void norm_rows_f16(const float* __restrict__ in, _Float16* __restrict__ out,
                              int n_rows) {
  int gt   = blockIdx.x * blockDim.x + threadIdx.x;
  int wave = gt >> 5;
  int lane = gt & 31;
  if (wave >= n_rows) return;
  float2 v = ((const float2*)(in + (long)wave * DIM))[lane];
  float ss = v.x * v.x + v.y * v.y;
  #pragma unroll
  for (int off = 16; off > 0; off >>= 1) ss += __shfl_xor(ss, off, 32);
  float scale = (ss > 1.0f) ? rsqrtf(ss) : 1.0f;  // 1/max(||row||,1)
  _Float16 o2[2] = {(_Float16)(v.x * scale), (_Float16)(v.y * scale)};
  *(__attribute__((ext_vector_type(2))) _Float16*)(out + (long)wave * DIM + 2 * lane) =
      *(__attribute__((ext_vector_type(2))) _Float16*)o2;
}

// ---------------- WMMA GEMM: H[M,N](f32) = X[M,K](f16) @ Wt[N,K](f16)^T ----------
// One wave per 16x16 output tile; K stepped 32 per v_wmma_f32_16x16x32_f16.
// Fragments assembled from contiguous b128 loads (ISA 7.12.2 wave32 layouts).
// Requires M % 16 == 0 (holds: 100000 = 6250*16).
template <int K, int N>
__global__ void gemm_wmma(const _Float16* __restrict__ X, const _Float16* __restrict__ Wt,
                          float* __restrict__ H, int M) {
  constexpr int NT = N / 16;
  int gt    = blockIdx.x * blockDim.x + threadIdx.x;
  int wave  = gt >> 5;
  int lane  = gt & 31;
  int mtile = wave / NT;
  int ntile = wave - mtile * NT;
  if (mtile * 16 >= M) return;                 // wave-uniform: EXEC stays all-1s

  int hi  = lane >> 4;                         // wave half
  int l15 = lane & 15;
  const _Float16* Xrow = X  + (long)(mtile * 16 + l15) * K;   // A row (M = l15)
  const _Float16* Wrow = Wt + (long)(ntile * 16 + l15) * K;   // B col (N = l15)

  v8f c = {};
  #pragma unroll
  for (int k0 = 0; k0 < K; k0 += 32) {
    // A: lane half hi -> K base k0+8*hi; elems 0-7 = K+0..7, elems 8-15 = K+16..23
    v8h a0 = *(const v8h*)(Xrow + k0 + hi * 8);
    v8h a1 = *(const v8h*)(Xrow + k0 + hi * 8 + 16);
    // B: lane half hi -> K base k0+8*hi; elems 0-7 = K+0..7, elems 8-15 = K+16..23 (Wt pre-transposed)
    v8h b0 = *(const v8h*)(Wrow + k0 + hi * 8);
    v8h b1 = *(const v8h*)(Wrow + k0 + hi * 8 + 16);
    v16h a = __builtin_shufflevector(a0, a1, 0,1,2,3,4,5,6,7,8,9,10,11,12,13,14,15);
    v16h b = __builtin_shufflevector(b0, b1, 0,1,2,3,4,5,6,7,8,9,10,11,12,13,14,15);
    // (neg_a, A, neg_b, B, c_mod, C, reuse_a, reuse_b)
    c = __builtin_amdgcn_wmma_f32_16x16x32_f16(false, a, false, b, (short)0, c,
                                               false, false);
  }
  // D: VGPR g -> row mtile*16 + hi*8 + g, col ntile*16 + l15 (16-lane coalesced)
  float* Hout = H + (long)(mtile * 16 + hi * 8) * N + ntile * 16 + l15;
  #pragma unroll
  for (int g = 0; g < 8; ++g) Hout[(long)g * N] = c[g];
}

// ---------------- edge scatter-add: agg[dst] += h[src] * dinv[src]*dinv[dst] ----
// One thread per (edge, 4-col chunk); wave covers contiguous columns of an edge
// -> coalesced float4 gathers (L2-resident) and contiguous f32 atomics.
__global__ void edge_agg(const float* __restrict__ H, const float* __restrict__ dinv,
                         const int* __restrict__ src, const int* __restrict__ dst,
                         float* __restrict__ agg, int E, int dim, int c4shift) {
  long t = (long)blockIdx.x * blockDim.x + threadIdx.x;
  int cols4 = 1 << c4shift;
  long total = (long)E * cols4;
  if (t >= total) return;
  int e  = (int)(t >> c4shift);
  int c4 = ((int)t & (cols4 - 1)) << 2;
  int s = src[e], d = dst[e];
  float coef = dinv[s] * dinv[d];
  const float4 v = *(const float4*)(H + (long)s * dim + c4);
  float* ap = agg + (long)d * dim + c4;
  atomicAdd(ap + 0, v.x * coef);
  atomicAdd(ap + 1, v.y * coef);
  atomicAdd(ap + 2, v.z * coef);
  atomicAdd(ap + 3, v.w * coef);
}

// ---------------- out = agg + h*dinv^2 + bias (, relu) -------------------------
// Writes f16 (next-layer GEMM operand) if out16 != null, else f32 in place.
__global__ void finalize(float* __restrict__ agg, const float* __restrict__ H,
                         const float* __restrict__ dinv, const float* __restrict__ bias,
                         long total, int dimshift, int do_relu,
                         _Float16* __restrict__ out16) {
  long t = (long)blockIdx.x * blockDim.x + threadIdx.x;
  if (t >= total) return;
  int node = (int)(t >> dimshift);
  int col  = (int)t & ((1 << dimshift) - 1);
  float di = dinv[node];
  float v = agg[t] + H[t] * di * di + bias[col];
  if (do_relu) v = fmaxf(v, 0.0f);
  if (out16) out16[t] = (_Float16)v;
  else       agg[t]   = v;
}

// ---------------- final gather + renorm(users) + dot + sigmoid ----------------
__global__ void predict(const int* __restrict__ u, const int* __restrict__ it,
                        const float* __restrict__ users_emb, const float* __restrict__ x2,
                        float* __restrict__ out, int B) {
  int gt   = blockIdx.x * blockDim.x + threadIdx.x;
  int wave = gt >> 5;
  int lane = gt & 31;
  if (wave >= B) return;
  int uu = u[wave], ii = it[wave];
  float2 uv = ((const float2*)(users_emb + (long)uu * DIM))[lane];
  float ss = uv.x * uv.x + uv.y * uv.y;
  #pragma unroll
  for (int off = 16; off > 0; off >>= 1) ss += __shfl_xor(ss, off, 32);
  float scale = (ss > 1.0f) ? rsqrtf(ss) : 1.0f;
  float2 xv = ((const float2*)(x2 + (long)ii * DIM))[lane];
  float dot = (uv.x * scale) * xv.x + (uv.y * scale) * xv.y;
  #pragma unroll
  for (int off = 16; off > 0; off >>= 1) dot += __shfl_xor(dot, off, 32);
  if (lane == 0) out[wave] = 1.0f / (1.0f + __expf(-dot));
}

// ---------------- host launcher ----------------

extern "C" void kernel_launch(void* const* d_in, const int* in_sizes, int n_in,
                              void* d_out, int out_size, void* d_ws, size_t ws_size,
                              hipStream_t stream) {
  const int*   u         = (const int*)d_in[0];
  const int*   ii        = (const int*)d_in[1];
  const int*   edges     = (const int*)d_in[2];
  const float* users_emb = (const float*)d_in[3];
  const float* item_emb  = (const float*)d_in[4];
  const float* W1        = (const float*)d_in[5];
  const float* b1        = (const float*)d_in[6];
  const float* W2        = (const float*)d_in[7];
  const float* b2        = (const float*)d_in[8];

  const int B       = in_sizes[0];
  const int E       = in_sizes[2] / 2;
  const int n_items = in_sizes[4] / DIM;
  const int HID     = 128;                       // b1 size (template constant)
  const int hshift  = 7;                         // log2(HID)

  const int* src = edges;
  const int* dst = edges + E;

  // workspace layout (f16 block first, then f32 blocks, then dinv)
  _Float16* X16 = (_Float16*)d_ws;                          // n_items*HID halves
  _Float16* Wt1 = X16 + (long)n_items * HID;                // 128*64 halves
  _Float16* Wt2 = Wt1 + (long)HID * DIM;                    // 64*128 halves
  float*    H   = (float*)(Wt2 + (long)DIM * HID);          // n_items*HID f32
  float*    AGG = H + (long)n_items * HID;                  // n_items*HID f32
  float*    DINV= AGG + (long)n_items * HID;                // n_items f32

  auto nb = [](long n) { return (unsigned)((n + TPB - 1) / TPB); };

  // degrees -> dinv (depends only on dst; shared by both layers)
  fill_zero  <<<nb(n_items), TPB, 0, stream>>>(DINV, (long)n_items);
  deg_count  <<<nb(E), TPB, 0, stream>>>(dst, DINV, E);
  deg_to_dinv<<<nb(n_items), TPB, 0, stream>>>(DINV, n_items);

  // weights -> f16 [N,K] for B-fragment b128 loads
  transpose_to_f16<<<nb((long)DIM * HID), TPB, 0, stream>>>(W1, Wt1, DIM, HID);
  transpose_to_f16<<<nb((long)HID * DIM), TPB, 0, stream>>>(W2, Wt2, HID, DIM);

  // x = max_norm1(item_emb) -> f16 (GEMM-only operand)
  norm_rows_f16<<<nb((long)n_items * 32), TPB, 0, stream>>>(item_emb, X16, n_items);

  const long mtiles = (n_items + 15) / 16;

  // ---- layer 1: relu(GCNConv(x, W1, b1)) ----
  gemm_wmma<DIM, 128><<<nb(mtiles * (HID / 16) * 32), TPB, 0, stream>>>(X16, Wt1, H,
                                                                        n_items);
  fill_zero<<<nb((long)n_items * HID), TPB, 0, stream>>>(AGG, (long)n_items * HID);
  edge_agg <<<nb((long)E * (HID / 4)), TPB, 0, stream>>>(H, DINV, src, dst, AGG,
                                                         E, HID, hshift - 2);
  finalize <<<nb((long)n_items * HID), TPB, 0, stream>>>(AGG, H, DINV, b1,
                                                         (long)n_items * HID, hshift, 1,
                                                         X16 /* f16 out, stride HID */);

  // ---- layer 2: GCNConv(x1, W2, b2) ----
  // h2 -> AGG (free after finalize1), agg2 -> H (free after finalize1 read)
  gemm_wmma<128, DIM><<<nb(mtiles * (DIM / 16) * 32), TPB, 0, stream>>>(X16, Wt2, AGG,
                                                                        n_items);
  fill_zero<<<nb((long)n_items * DIM), TPB, 0, stream>>>(H, (long)n_items * DIM);
  edge_agg <<<nb((long)E * (DIM / 4)), TPB, 0, stream>>>(AGG, DINV, src, dst, H,
                                                         E, DIM, 4);
  finalize <<<nb((long)n_items * DIM), TPB, 0, stream>>>(H, AGG, DINV, b2,
                                                         (long)n_items * DIM, 6, 0,
                                                         (_Float16*)nullptr);

  // ---- gather + dot + sigmoid ----
  predict<<<nb((long)B * 32), TPB, 0, stream>>>(u, ii, users_emb, H, (float*)d_out, B);
}
